// MixHistogram_68959994904956
// MI455X (gfx1250) — compile-verified, matrix operand order checked by reading, not run on hardware
//
#include <hip/hip_runtime.h>

typedef unsigned int u32;
typedef __attribute__((ext_vector_type(16))) _Float16 v16h;
typedef __attribute__((ext_vector_type(8)))  float    v8f;

#define NPIX    (512 * 512)   // pixels per (b,c) plane
#define CHANS   3
#define BATCH   64
#define NPLANES (BATCH * CHANS)
#define THREADS 256
#define CHUNK   (NPIX / THREADS)   // 1024 elements per thread per pass
#define NBINS   256
#define CSTRIDE 257                // padded row stride (bank-conflict free)
#define LDS_U32 (NBINS * CSTRIDE + 2 * NBINS)
#define LDS_BYTES (LDS_U32 * 4)

static_assert(CHUNK * THREADS == NPIX, "chunking must cover plane exactly");
static_assert(LDS_BYTES <= 320 * 1024, "must fit CDNA5 320KB LDS");

// ---- order-preserving float <-> u32 key mapping (stable total order) ----
__device__ __forceinline__ u32 map_key(float f) {
  u32 u = __float_as_uint(f);
  return (u & 0x80000000u) ? ~u : (u | 0x80000000u);
}
__device__ __forceinline__ float unmap_key(u32 k) {
  u32 u = (k & 0x80000000u) ? (k ^ 0x80000000u) : ~k;
  return __uint_as_float(u);
}

// ---- stage 1: build sortable (key, idx) pairs for source planes ----
extern "C" __global__ void mh_init_src(const float* __restrict__ x,
                                       u32* __restrict__ key, u32* __restrict__ idx) {
  const int p = blockIdx.y;
  const int i = blockIdx.x * THREADS + threadIdx.x;
  const size_t g = (size_t)p * NPIX + i;
  key[g] = map_key(x[g]);
  idx[g] = (u32)i;
}

// ---- stage 3: build sortable keys for permuted template planes ----
extern "C" __global__ void mh_init_tmpl(const float* __restrict__ x,
                                        const int* __restrict__ perm,
                                        u32* __restrict__ tkey) {
  const int p = blockIdx.y;
  const int b = p / CHANS, c = p % CHANS;
  const int sp = perm[b] * CHANS + c;
  const int i = blockIdx.x * THREADS + threadIdx.x;
  tkey[(size_t)p * NPIX + i] = map_key(x[(size_t)sp * NPIX + i]);
}

// ---- one stable LSD radix pass over one plane (one workgroup per plane) ----
// Per-thread 256-bin histograms live in LDS (256 digits x 256 threads, stride 257).
// Digit totals are reduced on the matrix pipe: v_wmma_f32_16x16x32_f16 with an
// all-ones B computes 16 row-sums per wave per call (counts <= 1024 exact in f16).
extern "C" __global__ void mh_radix_pass(const u32* __restrict__ keyIn, u32* __restrict__ keyOut,
                                         const u32* __restrict__ idxIn, u32* __restrict__ idxOut,
                                         int shift, int withIdx) {
  extern __shared__ u32 lds[];
  u32* cnt   = lds;                       // [NBINS][CSTRIDE]
  u32* dtot  = lds + NBINS * CSTRIDE;     // [NBINS] digit totals
  u32* dbase = dtot + NBINS;              // [NBINS] exclusive digit bases

  const int t = threadIdx.x;
  const size_t pb = (size_t)blockIdx.x * NPIX;
  keyIn += pb; keyOut += pb;
  if (withIdx) { idxIn += pb; idxOut += pb; }

  for (int i = t; i < NBINS * CSTRIDE; i += THREADS) cnt[i] = 0;
  __syncthreads();

  // --- per-thread histogram over a contiguous chunk (stable ordering basis) ---
  const int base = t * CHUNK;
  for (int j = 0; j < CHUNK; ++j) {
    if ((j & 31) == 0) __builtin_prefetch(keyIn + base + j + 128, 0, 0); // global_prefetch_b8
    u32 k = keyIn[base + j];
    cnt[((k >> shift) & 255u) * CSTRIDE + t] += 1u;
  }
  __syncthreads();

  // --- digit totals via WMMA row-sum: dtot[d] = sum_t cnt[d][t] ---
  {
    const int wave = t >> 5;
    const int lane = t & 31;
    const int g    = lane >> 4;   // half-wave group (selects K sub-range)
    const int m    = lane & 15;   // matrix row within 16-row group
    v16h ones;
#pragma unroll
    for (int e = 0; e < 16; ++e) ones[e] = (_Float16)1.0f;
#pragma unroll
    for (int G = 0; G < 2; ++G) {              // each wave reduces 2 groups of 16 digit-rows
      const int d0 = (wave * 2 + G) * 16;
      v8f acc = {};
      for (int kb = 0; kb < 8; ++kb) {         // 8 K-blocks of 32 columns = 256 threads
        v16h a;
#pragma unroll
        for (int e = 0; e < 16; ++e) {
          // ISA 16-bit A-matrix 16x32 lane layout: lanes 0-15 hold K 0-7/16-23,
          // lanes 16-31 hold K 8-15/24-31.
          int kk  = (e < 8) ? (g * 8 + e) : (16 + g * 8 + (e - 8));
          int col = kb * 32 + kk;
          a[e] = (_Float16)(float)cnt[(d0 + m) * CSTRIDE + col];
        }
        acc = __builtin_amdgcn_wmma_f32_16x16x32_f16(false, a, false, ones,
                                                     (short)0, acc, false, false);
      }
      // C/D layout: VGPR r holds M = r + 8*(lane/16); every N column equal => take N==0
      if ((lane & 15) == 0) {
#pragma unroll
        for (int r = 0; r < 8; ++r)
          dtot[d0 + r + 8 * g] = (u32)(acc[r] + 0.5f);
      }
    }
  }
  __syncthreads();

  // --- exclusive scan of 256 digit totals (tiny, off critical path) ---
  if (t == 0) {
    u32 run = 0;
    for (int d = 0; d < NBINS; ++d) { u32 c = dtot[d]; dbase[d] = run; run += c; }
  }
  __syncthreads();

  // --- per-digit-row exclusive prefix over thread columns -> absolute bases ---
  {
    u32 run = dbase[t];
    u32* row = cnt + t * CSTRIDE;
    for (int c = 0; c < THREADS; ++c) { u32 v = row[c]; row[c] = run; run += v; }
  }
  __syncthreads();

  // --- stable scatter ---
  if (withIdx) {
    for (int j = 0; j < CHUNK; ++j) {
      u32 k = keyIn[base + j];
      u32 d = (k >> shift) & 255u;
      u32 pos = cnt[d * CSTRIDE + t]++;
      keyOut[pos] = k;
      idxOut[pos] = idxIn[base + j];
    }
  } else {
    for (int j = 0; j < CHUNK; ++j) {
      u32 k = keyIn[base + j];
      u32 d = (k >> shift) & 255u;
      u32 pos = cnt[d * CSTRIDE + t]++;
      keyOut[pos] = k;
    }
  }
}

// ---- stage 5: rank-matched blend. out[order[k]] = x + (tmpl_sorted[k]-x)*(1-lmda) ----
extern "C" __global__ void mh_finalize(const float* __restrict__ x,
                                       const float* __restrict__ lmda,
                                       const u32* __restrict__ sidxSorted,
                                       const u32* __restrict__ tkeySorted,
                                       float* __restrict__ out) {
  const int p = blockIdx.y;
  const int k = blockIdx.x * THREADS + threadIdx.x;
  const size_t pb = (size_t)p * NPIX;
  const u32 pixel = sidxSorted[pb + k];
  const float tv  = unmap_key(tkeySorted[pb + k]);
  const float xv  = x[pb + pixel];
  const float l   = lmda[p / CHANS];
  out[pb + pixel] = xv + (tv - xv) * (1.0f - l);
}

extern "C" void kernel_launch(void* const* d_in, const int* in_sizes, int n_in,
                              void* d_out, int out_size, void* d_ws, size_t ws_size,
                              hipStream_t stream) {
  (void)in_sizes; (void)n_in; (void)out_size; (void)ws_size;
  const float* x    = (const float*)d_in[0];
  const float* lmda = (const float*)d_in[1];
  const int*   perm = (const int*)d_in[2];
  float*       out  = (float*)d_out;

  const size_t PN = (size_t)NPLANES * NPIX;
  u32* skeyA = (u32*)d_ws;        // sortable keys, ping
  u32* skeyB = skeyA + PN;        // sortable keys, pong
  u32* sidxA = skeyB + PN;        // source indices, ping
  u32* sidxB = sidxA + PN;        // source indices, pong

  hipFuncSetAttribute((const void*)mh_radix_pass,
                      hipFuncAttributeMaxDynamicSharedMemorySize, (int)LDS_BYTES);

  dim3 blk(THREADS);
  dim3 grdElem(NPIX / THREADS, NPLANES);

  // source argsort: 4 stable passes on (key, idx); ends back in (skeyA, sidxA)
  mh_init_src<<<grdElem, blk, 0, stream>>>(x, skeyA, sidxA);
  mh_radix_pass<<<NPLANES, blk, LDS_BYTES, stream>>>(skeyA, skeyB, sidxA, sidxB,  0, 1);
  mh_radix_pass<<<NPLANES, blk, LDS_BYTES, stream>>>(skeyB, skeyA, sidxB, sidxA,  8, 1);
  mh_radix_pass<<<NPLANES, blk, LDS_BYTES, stream>>>(skeyA, skeyB, sidxA, sidxB, 16, 1);
  mh_radix_pass<<<NPLANES, blk, LDS_BYTES, stream>>>(skeyB, skeyA, sidxB, sidxA, 24, 1);

  // template sort: reuse key buffers (sorted src keys are dead; xv re-read from x).
  // Starts in skeyB, 4 passes end back in skeyB; sidxA untouched.
  mh_init_tmpl<<<grdElem, blk, 0, stream>>>(x, perm, skeyB);
  mh_radix_pass<<<NPLANES, blk, LDS_BYTES, stream>>>(skeyB, skeyA, nullptr, nullptr,  0, 0);
  mh_radix_pass<<<NPLANES, blk, LDS_BYTES, stream>>>(skeyA, skeyB, nullptr, nullptr,  8, 0);
  mh_radix_pass<<<NPLANES, blk, LDS_BYTES, stream>>>(skeyB, skeyA, nullptr, nullptr, 16, 0);
  mh_radix_pass<<<NPLANES, blk, LDS_BYTES, stream>>>(skeyA, skeyB, nullptr, nullptr, 24, 0);

  mh_finalize<<<grdElem, blk, 0, stream>>>(x, lmda, sidxA, skeyB, out);
}
